// WaveAttention_50861002719393
// MI455X (gfx1250) — matrix operand order repaired and checked by reading, not compiled, not run on hardware
//
#include <hip/hip_runtime.h>
#include <hip/hip_bf16.h>
#include <math.h>

typedef __attribute__((ext_vector_type(16))) _Float16 v16h;
typedef __attribute__((ext_vector_type(8)))  _Float16 v8h;
typedef __attribute__((ext_vector_type(8)))  float    v8f;

#define EMB      4096
#define HEADS    8
#define HEAD_DIM 512
#define BATCH    8
#define SEQ      512
#define BH       (BATCH * HEADS)
#define MROWS    (BATCH * SEQ)

#define BM 128
#define BN 128
#define BK 32
#define LDT 40          // LDS tile row stride in halves (32 data + 8 pad = 80B)
#define NTHREADS 256    // 8 waves (wave32)

// ---------------------------------------------------------------------------
// WMMA wrapper: D = A(16x32 f16) * B(32x16 f16) + C(f32)
// ---------------------------------------------------------------------------
__device__ __forceinline__ v8f wmma_f16(v16h a, v16h b, v8f c) {
  return __builtin_amdgcn_wmma_f32_16x16x32_f16(
      /*neg_a=*/false, a, /*neg_b=*/false, b,
      /*c_mod=*/(short)0, c, /*reuse_a=*/false, /*reuse_b=*/false);
}

union HalfVec { v16h v; v8h h[2]; };

// A fragment (16x32 f16), CDNA5 layout: lane l -> row m=l&15,
// VGPR0-3 = K=kb..kb+7, VGPR4-7 = K=kb+16..kb+23, kb = (l>=16)*8.
__device__ __forceinline__ v16h frag_a(const _Float16* lds, int row, int lane) {
  int m  = lane & 15;
  int kb = (lane >> 4) << 3;
  const _Float16* p = lds + (row + m) * LDT + kb;
  HalfVec u;
  u.h[0] = *(const v8h*)(p);
  u.h[1] = *(const v8h*)(p + 16);
  return u.v;
}

// B fragment (32x16 f16): lane l -> col n=l&15, VGPR0-3 = K=kb..kb+7,
// VGPR4-7 = K=kb+16..kb+23, kb = (l>=16)*8. LDS B tile stored [n][k].
__device__ __forceinline__ v16h frag_b(const _Float16* lds, int col, int lane) {
  int n  = lane & 15;
  int kb = (lane >> 4) << 3;
  const _Float16* p = lds + (col + n) * LDT + kb;
  HalfVec u;
  u.h[0] = *(const v8h*)(p);
  u.h[1] = *(const v8h*)(p + 16);
  return u.v;
}

// ---------------------------------------------------------------------------
// CDNA5 async copy: GLOBAL_LOAD_ASYNC_TO_LDS_B128 (ASYNCcnt-tracked DMA into
// LDS, §15.18.3 op 98). Generic LDS pointer truncates to the 32-bit LDS
// offset.
// ---------------------------------------------------------------------------
__device__ __forceinline__ void async_copy_b128(void* lds_dst, const void* gsrc) {
  unsigned lds_off = (unsigned)(unsigned long long)lds_dst;
  asm volatile("global_load_async_to_lds_b128 %0, %1, off"
               :: "v"(lds_off), "v"(gsrc)
               : "memory");
}

__device__ __forceinline__ void wait_async0() {
#if __has_builtin(__builtin_amdgcn_s_wait_asynccnt)
  __builtin_amdgcn_s_wait_asynccnt(0);
#else
  asm volatile("s_wait_asynccnt 0x0" ::: "memory");
#endif
}

// Stage a 128x32 f16 tile global->LDS via async DMA. Thread t: row t>>1,
// halves (t&1)*16..+15 -> two b128 async copies per thread.
__device__ __forceinline__ void stage_tile_async(_Float16* lds, const _Float16* src,
                                                 size_t ld, int row0, int k0, int tid) {
  int r = tid >> 1;
  int c = (tid & 1) << 4;
  const _Float16* s = src + (size_t)(row0 + r) * ld + (size_t)(k0 + c);
  _Float16* d = lds + r * LDT + c;
  async_copy_b128(d,     s);
  async_copy_b128(d + 8, s + 8);
}

// ---------------------------------------------------------------------------
// Kernel 0: bulk f32 -> f16 convert (8 elements / thread)
// ---------------------------------------------------------------------------
__global__ __launch_bounds__(256) void cvt_f16_kernel(const float* __restrict__ s,
                                                      _Float16* __restrict__ d) {
  size_t i = ((size_t)blockIdx.x * 256 + threadIdx.x) * 8;
  float4 f0 = ((const float4*)(s + i))[0];
  float4 f1 = ((const float4*)(s + i))[1];
  v8h h;
  h[0] = (_Float16)f0.x; h[1] = (_Float16)f0.y; h[2] = (_Float16)f0.z; h[3] = (_Float16)f0.w;
  h[4] = (_Float16)f1.x; h[5] = (_Float16)f1.y; h[6] = (_Float16)f1.z; h[7] = (_Float16)f1.w;
  *(v8h*)(d + i) = h;
}

// ---------------------------------------------------------------------------
// Generic f16 GEMM (async double-buffered): C = A(f16)[M,K] @ W(f16)[N,K]^T.
//   MODE 0: C f32 = acc + bias[n]                        (Q/K projection)
//   MODE 1: vt f16 transposed-store = acc + bias[n]      (V projection)
//   MODE 2: C f32 = acc + bias[n] + resid[m,n]           (output projection)
// ---------------------------------------------------------------------------
template <int MODE>
__global__ __launch_bounds__(NTHREADS) void gemm_f16_kernel(
    const _Float16* __restrict__ A, const _Float16* __restrict__ W,
    const float* __restrict__ bias, const float* __restrict__ resid,
    float* __restrict__ Cf, _Float16* __restrict__ Ch, int Kdim) {
  __shared__ __align__(32) _Float16 ldsA[2][BM * LDT];
  __shared__ __align__(32) _Float16 ldsB[2][BN * LDT];
  int tid = threadIdx.x, lane = tid & 31, wid = tid >> 5;
  int wm = (wid & 3) * 32, wn = (wid >> 2) * 64;
  int row0 = blockIdx.y * BM, col0 = blockIdx.x * BN;
  v8f acc[2][4] = {};

  stage_tile_async(ldsA[0], A, Kdim, row0, 0, tid);
  stage_tile_async(ldsB[0], W, Kdim, col0, 0, tid);

  for (int k0 = 0, it = 0; k0 < Kdim; k0 += BK, ++it) {
    int p = it & 1;
    wait_async0();        // my async writes into buffer p have landed
    __syncthreads();      // everyone's landed; prior reads of p^1 consumed
    if (k0 + BK < Kdim) { // fill the other buffer underneath the WMMAs
      stage_tile_async(ldsA[p ^ 1], A, Kdim, row0, k0 + BK, tid);
      stage_tile_async(ldsB[p ^ 1], W, Kdim, col0, k0 + BK, tid);
      if (k0 + 2 * BK < Kdim)
        __builtin_prefetch(A + (size_t)(row0 + (tid >> 1)) * Kdim + (k0 + 2 * BK), 0, 1);
    }
    v16h af[2], bf[4];
    for (int i = 0; i < 2; ++i) af[i] = frag_a(ldsA[p], wm + i * 16, lane);
    for (int j = 0; j < 4; ++j) bf[j] = frag_b(ldsB[p], wn + j * 16, lane);
    for (int i = 0; i < 2; ++i)
      for (int j = 0; j < 4; ++j)
        acc[i][j] = wmma_f16(af[i], bf[j], acc[i][j]);
  }

  int cn = lane & 15, rb = (lane >> 4) << 3;
  for (int i = 0; i < 2; ++i)
    for (int j = 0; j < 4; ++j) {
      int gn = col0 + wn + j * 16 + cn;
      float bv = bias[gn];
      for (int v = 0; v < 8; ++v) {
        int gm = row0 + wm + i * 16 + rb + v;
        float r = acc[i][j][v] + bv;
        if (MODE == 0) {
          Cf[(size_t)gm * EMB + gn] = r;
        } else if (MODE == 1) {
          // vt[((b*H+h)*D + d)*S + s] ; gm=(b,s), gn=(h,d)
          int b = gm >> 9, s = gm & (SEQ - 1);
          int h = gn >> 9, d = gn & (HEAD_DIM - 1);
          Ch[((size_t)(b * HEADS + h) * HEAD_DIM + d) * SEQ + s] = (_Float16)r;
        } else {
          size_t off = (size_t)gm * EMB + gn;
          Cf[off] = r + resid[off];
        }
      }
    }
}

// ---------------------------------------------------------------------------
// Kernel 2: g_q[b,e] = sqrt(sum_s Q[b,s,e]^2)
// ---------------------------------------------------------------------------
__global__ __launch_bounds__(256) void colnorm_q_kernel(const float* __restrict__ Q,
                                                        float* __restrict__ gq) {
  int t = blockIdx.x * 256 + threadIdx.x;   // b*EMB + e
  int b = t >> 12;
  int e = t & (EMB - 1);
  const float* p = Q + (size_t)b * SEQ * EMB + e;
  float s = 0.f;
  for (int i = 0; i < SEQ; ++i) { float v = p[(size_t)i * EMB]; s += v * v; }
  gq[t] = sqrtf(s);
}

// ---------------------------------------------------------------------------
// Kernel 3: g_k[b,h,t] = sqrt(sum_d K[b,t,h*D+d]^2)  (wave32 per row)
// ---------------------------------------------------------------------------
__global__ __launch_bounds__(256) void rownorm_k_kernel(const float* __restrict__ Kf,
                                                        float* __restrict__ gk) {
  int gw = (blockIdx.x * 256 + threadIdx.x) >> 5;  // row id: (b,h,t)
  int lane = threadIdx.x & 31;
  int b = gw / (HEADS * SEQ);
  int rem = gw - b * (HEADS * SEQ);
  int h = rem / SEQ;
  int t = rem - h * SEQ;
  const float* p = Kf + (size_t)(b * SEQ + t) * EMB + h * HEAD_DIM;
  float s = 0.f;
  for (int i = lane; i < HEAD_DIM; i += 32) { float v = p[i]; s += v * v; }
  for (int off = 16; off; off >>= 1) s += __shfl_xor(s, off, 32);
  if (lane == 0) gk[gw] = sqrtf(s);
}

// ---------------------------------------------------------------------------
// Kernel 4: build packed complex operands (f16):
//   qpack = [qr|qi], qpack2 = [qi|-qr], kpack = [kr|ki]   (K = 1024)
// ---------------------------------------------------------------------------
__device__ __forceinline__ void wave_complex(float v, float g, float& re, float& im) {
  float r  = v / (g + 1e-8f);
  float st = sqrtf(fmaxf(0.f, fminf(1.f, 1.f - r * r)));
  float hy = sqrtf(r * r + st * st);      // ==1 for |r|<=1, |r| otherwise
  float inv = hy > 0.f ? 1.f / hy : 0.f;
  re = g * r * inv;
  im = g * st * inv;
}

__global__ __launch_bounds__(256) void pack_kernel(
    const float* __restrict__ Q, const float* __restrict__ Kf,
    const float* __restrict__ gq, const float* __restrict__ gk,
    _Float16* __restrict__ qpack, _Float16* __restrict__ qpack2,
    _Float16* __restrict__ kpack) {
  size_t idx = (size_t)blockIdx.x * 256 + threadIdx.x;  // over B*SEQ*EMB
  int e = (int)(idx & (EMB - 1));
  size_t bs = idx >> 12;
  int s = (int)(bs & (SEQ - 1));
  int b = (int)(bs >> 9);
  int h = e >> 9;
  int d = e & (HEAD_DIM - 1);
  size_t bh_row = (size_t)(b * HEADS + h) * SEQ;
  // Q -> (qr, qi), g_q indexed by (b, e)
  {
    float re, im;
    wave_complex(Q[idx], gq[b * EMB + e], re, im);
    _Float16* dst  = qpack  + (bh_row + s) * 1024;
    _Float16* dst2 = qpack2 + (bh_row + s) * 1024;
    dst[d]        = (_Float16)re;
    dst[512 + d]  = (_Float16)im;
    dst2[d]       = (_Float16)im;
    dst2[512 + d] = (_Float16)(-re);
  }
  // K -> (kr, ki), g_k indexed by d (reference broadcast quirk: D == S)
  {
    float re, im;
    wave_complex(Kf[idx], gk[(b * HEADS + h) * SEQ + d], re, im);
    _Float16* dst = kpack + (bh_row + s) * 1024;
    dst[d] = (_Float16)re;
    dst[512 + d] = (_Float16)im;
  }
}

// ---------------------------------------------------------------------------
// Kernel 5: attention scores per (b,h): real = [qr|qi].[kr|ki],
//   imag = [qi|-qr].[kr|ki] (K=1024), scores = sqrt(real^2+imag^2).
// ---------------------------------------------------------------------------
__global__ __launch_bounds__(NTHREADS) void attn_score_kernel(
    const _Float16* __restrict__ qpack, const _Float16* __restrict__ qpack2,
    const _Float16* __restrict__ kpack, float* __restrict__ scores) {
  __shared__ __align__(32) _Float16 ldsAr[2][BM * LDT];
  __shared__ __align__(32) _Float16 ldsAi[2][BM * LDT];
  __shared__ __align__(32) _Float16 ldsB [2][BN * LDT];
  int bh = blockIdx.z;
  const _Float16* Qr = qpack  + (size_t)bh * SEQ * 1024;
  const _Float16* Qi = qpack2 + (size_t)bh * SEQ * 1024;
  const _Float16* Kp = kpack  + (size_t)bh * SEQ * 1024;
  float* Sc = scores + (size_t)bh * SEQ * SEQ;
  int tid = threadIdx.x, lane = tid & 31, wid = tid >> 5;
  int wm = (wid & 3) * 32, wn = (wid >> 2) * 64;
  int row0 = blockIdx.y * BM, col0 = blockIdx.x * BN;
  v8f accR[2][4] = {}, accI[2][4] = {};

  stage_tile_async(ldsAr[0], Qr, 1024, row0, 0, tid);
  stage_tile_async(ldsAi[0], Qi, 1024, row0, 0, tid);
  stage_tile_async(ldsB [0], Kp, 1024, col0, 0, tid);

  for (int k0 = 0, it = 0; k0 < 1024; k0 += BK, ++it) {
    int p = it & 1;
    wait_async0();
    __syncthreads();
    if (k0 + BK < 1024) {
      stage_tile_async(ldsAr[p ^ 1], Qr, 1024, row0, k0 + BK, tid);
      stage_tile_async(ldsAi[p ^ 1], Qi, 1024, row0, k0 + BK, tid);
      stage_tile_async(ldsB [p ^ 1], Kp, 1024, col0, k0 + BK, tid);
    }
    v16h ar[2], ai[2], bf[4];
    for (int i = 0; i < 2; ++i) {
      ar[i] = frag_a(ldsAr[p], wm + i * 16, lane);
      ai[i] = frag_a(ldsAi[p], wm + i * 16, lane);
    }
    for (int j = 0; j < 4; ++j) bf[j] = frag_b(ldsB[p], wn + j * 16, lane);
    for (int i = 0; i < 2; ++i)
      for (int j = 0; j < 4; ++j) {
        accR[i][j] = wmma_f16(ar[i], bf[j], accR[i][j]);
        accI[i][j] = wmma_f16(ai[i], bf[j], accI[i][j]);
      }
  }
  int cn = lane & 15, rb = (lane >> 4) << 3;
  for (int i = 0; i < 2; ++i)
    for (int j = 0; j < 4; ++j)
      for (int v = 0; v < 8; ++v) {
        int gm = row0 + wm + i * 16 + rb + v;
        int gn = col0 + wn + j * 16 + cn;
        float re = accR[i][j][v], im = accI[i][j][v];
        Sc[(size_t)gm * SEQ + gn] = sqrtf(re * re + im * im);
      }
}

// ---------------------------------------------------------------------------
// Kernel 6: masked, scaled softmax over t (wave32 per row), probs -> f16
// ---------------------------------------------------------------------------
__global__ __launch_bounds__(256) void softmax_kernel(
    const float* __restrict__ scores, const float* __restrict__ amask,
    _Float16* __restrict__ probs) {
  int gw = (blockIdx.x * 256 + threadIdx.x) >> 5;  // row: ((b*H)+h)*S + s
  int lane = threadIdx.x & 31;
  int b = gw >> 12;  // / (H*S)
  const float* row = scores + (size_t)gw * SEQ;
  const float scale = 0.044194173824159216f;  // 1/sqrt(512)
  float vals[16];
  float mx = -__builtin_inff();
  for (int i = 0; i < 16; ++i) {
    int t = lane + i * 32;
    float m = amask[b * SEQ + t];
    float v = (m == 0.f) ? -__builtin_inff() : row[t] * scale;
    vals[i] = v;
    mx = fmaxf(mx, v);
  }
  for (int off = 16; off; off >>= 1) mx = fmaxf(mx, __shfl_xor(mx, off, 32));
  float sum = 0.f;
  for (int i = 0; i < 16; ++i) { float e = __expf(vals[i] - mx); vals[i] = e; sum += e; }
  for (int off = 16; off; off >>= 1) sum += __shfl_xor(sum, off, 32);
  float inv = 1.f / sum;
  _Float16* prow = probs + (size_t)gw * SEQ;
  for (int i = 0; i < 16; ++i) prow[lane + i * 32] = (_Float16)(vals[i] * inv);
}

// ---------------------------------------------------------------------------
// Kernel 7: ctx = probs @ V  per (b,h): A=probs[s][t], B=vt[d][t], K=512.
// ---------------------------------------------------------------------------
__global__ __launch_bounds__(NTHREADS) void attn_pv_kernel(
    const _Float16* __restrict__ probs, const _Float16* __restrict__ vt,
    _Float16* __restrict__ ctx) {
  __shared__ __align__(32) _Float16 ldsA[2][BM * LDT];
  __shared__ __align__(32) _Float16 ldsB[2][BN * LDT];
  int bh = blockIdx.z;
  int b = bh / HEADS, h = bh - b * HEADS;
  const _Float16* P = probs + (size_t)bh * SEQ * SEQ;
  const _Float16* V = vt + (size_t)bh * HEAD_DIM * SEQ;
  int tid = threadIdx.x, lane = tid & 31, wid = tid >> 5;
  int wm = (wid & 3) * 32, wn = (wid >> 2) * 64;
  int row0 = blockIdx.y * BM, col0 = blockIdx.x * BN;
  v8f acc[2][4] = {};

  stage_tile_async(ldsA[0], P, SEQ, row0, 0, tid);
  stage_tile_async(ldsB[0], V, SEQ, col0, 0, tid);

  for (int k0 = 0, it = 0; k0 < SEQ; k0 += BK, ++it) {
    int p = it & 1;
    wait_async0();
    __syncthreads();
    if (k0 + BK < SEQ) {
      stage_tile_async(ldsA[p ^ 1], P, SEQ, row0, k0 + BK, tid);
      stage_tile_async(ldsB[p ^ 1], V, SEQ, col0, k0 + BK, tid);
    }
    v16h af[2], bf[4];
    for (int i = 0; i < 2; ++i) af[i] = frag_a(ldsA[p], wm + i * 16, lane);
    for (int j = 0; j < 4; ++j) bf[j] = frag_b(ldsB[p], wn + j * 16, lane);
    for (int i = 0; i < 2; ++i)
      for (int j = 0; j < 4; ++j)
        acc[i][j] = wmma_f16(af[i], bf[j], acc[i][j]);
  }
  int cn = lane & 15, rb = (lane >> 4) << 3;
  for (int i = 0; i < 2; ++i)
    for (int j = 0; j < 4; ++j)
      for (int v = 0; v < 8; ++v) {
        int gm = row0 + wm + i * 16 + rb + v;   // s
        int gn = col0 + wn + j * 16 + cn;       // d
        ctx[(size_t)(b * SEQ + gm) * EMB + h * HEAD_DIM + gn] = (_Float16)acc[i][j][v];
      }
}

// ---------------------------------------------------------------------------
// Kernel 9: LayerNorm over last dim (block per row)
// ---------------------------------------------------------------------------
__global__ __launch_bounds__(256) void layernorm_kernel(
    const float* __restrict__ y, const float* __restrict__ w,
    const float* __restrict__ bparm, float* __restrict__ out) {
  __shared__ float sA[8], sB[8];
  int row = blockIdx.x;
  const float* p = y + (size_t)row * EMB;
  float s = 0.f, s2 = 0.f;
  for (int i = threadIdx.x; i < EMB; i += 256) {
    float v = p[i];
    s += v; s2 += v * v;
  }
  for (int off = 16; off; off >>= 1) { s += __shfl_xor(s, off, 32); s2 += __shfl_xor(s2, off, 32); }
  int lane = threadIdx.x & 31, wid = threadIdx.x >> 5;
  if (lane == 0) { sA[wid] = s; sB[wid] = s2; }
  __syncthreads();
  float ts = 0.f, ts2 = 0.f;
  for (int i = 0; i < 8; ++i) { ts += sA[i]; ts2 += sB[i]; }
  float mu = ts * (1.f / EMB);
  float var = ts2 * (1.f / EMB) - mu * mu;
  float rstd = rsqrtf(var + 1e-5f);
  for (int i = threadIdx.x; i < EMB; i += 256) {
    out[(size_t)row * EMB + i] = (p[i] - mu) * rstd * w[i] + bparm[i];
  }
}

// ---------------------------------------------------------------------------
// Launch pipeline (peak ws ~416.25 MB, region reuse is stream-ordered):
//   [0,64)    Qf f32          -> later scores f32 / yb overlap
//   [64,128)  Kf f32          -> later probs f16 + ctx f16
//   [128,192) qp f16
//   [192,256) kp f16
//   [256,288) vt f16
//   [288,352) qp2 f16
//   [352,384) xh f16
//   [384,416) Wh f16 (Wq/Wk/Wv/Wo converted sequentially into this slot)
//   [416,..)  gq, gk
// ---------------------------------------------------------------------------
extern "C" void kernel_launch(void* const* d_in, const int* in_sizes, int n_in,
                              void* d_out, int out_size, void* d_ws, size_t ws_size,
                              hipStream_t stream) {
  (void)in_sizes; (void)n_in; (void)out_size; (void)ws_size;
  const float* x   = (const float*)d_in[0];
  const float* am  = (const float*)d_in[1];
  const float* Wq  = (const float*)d_in[2];
  const float* bq  = (const float*)d_in[3];
  const float* Wk  = (const float*)d_in[4];
  const float* bk  = (const float*)d_in[5];
  const float* Wv  = (const float*)d_in[6];
  const float* bv  = (const float*)d_in[7];
  const float* Wo  = (const float*)d_in[8];
  const float* bo  = (const float*)d_in[9];
  const float* lnw = (const float*)d_in[10];
  const float* lnb = (const float*)d_in[11];
  float* out = (float*)d_out;

  char* ws = (char*)d_ws;
  const size_t MB = 1ull << 20;
  float*    Qf  = (float*)(ws + 0);            // 64 MB
  float*    Kf  = (float*)(ws + 64 * MB);      // 64 MB
  _Float16* qp  = (_Float16*)(ws + 128 * MB);  // 64 MB
  _Float16* kp  = (_Float16*)(ws + 192 * MB);  // 64 MB
  _Float16* vt  = (_Float16*)(ws + 256 * MB);  // 32 MB
  _Float16* qp2 = (_Float16*)(ws + 288 * MB);  // 64 MB
  _Float16* xh  = (_Float16*)(ws + 352 * MB);  // 32 MB
  _Float16* Wh  = (_Float16*)(ws + 384 * MB);  // 32 MB (reused slot)
  float*    gq  = (float*)(ws + 416 * MB);              // 128 KB
  float*    gk  = (float*)(ws + 416 * MB + 256 * 1024); // 128 KB
  // phase-2 reuse (each region dead before its overwrite, stream-ordered)
  float*    scores = (float*)(ws + 0);           // 64 MB (over Qf)
  _Float16* probs  = (_Float16*)(ws + 64 * MB);  // 32 MB (over Kf lo)
  _Float16* ctx    = (_Float16*)(ws + 96 * MB);  // 32 MB (over Kf hi)
  float*    yb     = (float*)(ws + 0);           // 64 MB (over scores, dead)

  dim3 blk(NTHREADS);
  dim3 gproj(EMB / BN, MROWS / BM);        // 32 x 32
  dim3 gattn(SEQ / BN, SEQ / BM, BH);      // 4 x 4 x 64
  const int nW  = (EMB * EMB) / (256 * 8);   // cvt grid for 4096x4096
  const int nX  = nW;                        // x has same element count

  cvt_f16_kernel<<<nX, 256, 0, stream>>>(x, xh);

  cvt_f16_kernel<<<nW, 256, 0, stream>>>(Wq, Wh);
  gemm_f16_kernel<0><<<gproj, blk, 0, stream>>>(xh, Wh, bq, nullptr, Qf, nullptr, EMB);
  cvt_f16_kernel<<<nW, 256, 0, stream>>>(Wk, Wh);
  gemm_f16_kernel<0><<<gproj, blk, 0, stream>>>(xh, Wh, bk, nullptr, Kf, nullptr, EMB);
  cvt_f16_kernel<<<nW, 256, 0, stream>>>(Wv, Wh);
  gemm_f16_kernel<1><<<gproj, blk, 0, stream>>>(xh, Wh, bv, nullptr, nullptr, vt, EMB);

  colnorm_q_kernel<<<(BATCH * EMB) / 256, 256, 0, stream>>>(Qf, gq);
  rownorm_k_kernel<<<(BATCH * HEADS * SEQ * 32) / 256, 256, 0, stream>>>(Kf, gk);
  pack_kernel<<<(int)(((size_t)BATCH * SEQ * EMB) / 256), 256, 0, stream>>>(
      Qf, Kf, gq, gk, qp, qp2, kp);

  attn_score_kernel<<<gattn, blk, 0, stream>>>(qp, qp2, kp, scores);
  softmax_kernel<<<(BATCH * HEADS * SEQ * 32) / 256, 256, 0, stream>>>(scores, am, probs);
  attn_pv_kernel<<<gattn, blk, 0, stream>>>(probs, vt, ctx);

  cvt_f16_kernel<<<nW, 256, 0, stream>>>(Wo, Wh);
  gemm_f16_kernel<2><<<gproj, blk, 0, stream>>>(ctx, Wh, bo, x, yb, nullptr, EMB);
  layernorm_kernel<<<MROWS, 256, 0, stream>>>(yb, lnw, lnb, out);
}
